// DecoderRNN_17179869836
// MI455X (gfx1250) — matrix operand order repaired and checked by reading, not compiled, not run on hardware
//
#include <hip/hip_runtime.h>

// ---------------------------------------------------------------------------
// DecoderRNN (image captioning LSTM). Attention collapses (single annotation:
// softmax over size-1 axis == 1, context == features), so the math is:
//   h0=feat@iHW+b, c0=feat@iCW+b
//   per t: gates=[emb_t|feat|h]@[W_ih|W_hh]^T + b_ih+b_hh ; LSTM pointwise ;
//          out_t = h_new @ fc_W + fc_b   (128x32000, K=256  <- dominant)
// Strategy: bf16 WMMA (v_wmma_f32_16x16x32_bf16) everywhere, weights packed
// once into the per-lane B-fragment layout, activations bf16 in one concat
// buffer. gates GEMM uses split-K x4 inside one launch (2048 waves) because
// at 512 waves it is pure L2-latency; fc GEMM caches A (all K=256) in regs.
// ---------------------------------------------------------------------------

#define B_   128
#define T_   32
#define V_   32000
#define E_   128
#define H_   256
#define F_   2048
#define KX   2432          // E + F + H  (concat GEMM K)
#define KXS  76            // KX / 32
#define KSPLIT 4           // split-K factor for gates (76 = 4*19)
#define KCH  19            // k-steps per split chunk
#define G4H  1024          // 4*H
#define KFS  8             // H / 32   (fc GEMM k-steps)
#define KIS  64            // F / 32   (init GEMM k-steps)
#define GBS  (B_ * G4H)    // 131072 floats per gates partial

typedef __attribute__((ext_vector_type(16))) __bf16 v16bf;
typedef __attribute__((ext_vector_type(8)))  __bf16 v8bf;
typedef __attribute__((ext_vector_type(8)))  float  v8f;

__device__ __forceinline__ v16bf cat8(v8bf lo, v8bf hi) {
  union { v16bf v; v8bf h[2]; } u;
  u.h[0] = lo; u.h[1] = hi;
  return u.v;
}

// A-fragment (16x32 bf16) from row-major: lane<16 -> K0..7 & K16..23,
// lane>=16 -> K8..15 & K24..31 (two 16B loads; lane K-offset folded into p).
__device__ __forceinline__ v16bf load_a(const __bf16* p) {
  return cat8(*(const v8bf*)p, *(const v8bf*)(p + 16));
}

// ---------------- weight packing (idempotent, every call) -------------------
// B-fragment for 16x16x32 bf16: lane l holds column N=l%16, K values
// (l<16 ? 0..7,16..23 : 8..15,24..31) within the k-step -> 16 bf16 per lane.
__global__ void pack_fc(const float* __restrict__ fcW, __bf16* __restrict__ out) {
  int idx = blockIdx.x * blockDim.x + threadIdx.x;     // 2000*8*32*16 = 8,192,000
  int e    = idx & 15;
  int lane = (idx >> 4) & 31;
  int ks   = (idx >> 9) & 7;
  int nt   = idx >> 12;
  int K = ks * 32 + ((lane >= 16) ? 8 : 0) + (e & 7) + ((e & 8) << 1);
  int N = nt * 16 + (lane & 15);
  out[idx] = (__bf16)fcW[(size_t)K * V_ + N];
}

__global__ void pack_gates(const float* __restrict__ Wih, const float* __restrict__ Whh,
                           __bf16* __restrict__ out) {
  int idx = blockIdx.x * blockDim.x + threadIdx.x;     // 64*76*32*16 = 2,490,368
  int e    = idx & 15;
  int lane = (idx >> 4) & 31;
  int rest = idx >> 9;
  int ks   = rest % KXS;
  int nt   = rest / KXS;
  int K = ks * 32 + ((lane >= 16) ? 8 : 0) + (e & 7) + ((e & 8) << 1);       // 0..2431
  int N = nt * 16 + (lane & 15);                       // 0..1023
  float w = (K < E_ + F_) ? Wih[(size_t)N * (E_ + F_) + K]
                          : Whh[(size_t)N * H_ + (K - (E_ + F_))];
  out[idx] = (__bf16)w;
}

// [initH_W | initC_W] : (2048 x 512), N<256 -> H, N>=256 -> C
__global__ void pack_init(const float* __restrict__ iHW, const float* __restrict__ iCW,
                          __bf16* __restrict__ out) {
  int idx = blockIdx.x * blockDim.x + threadIdx.x;     // 32*64*32*16 = 1,048,576
  int e    = idx & 15;
  int lane = (idx >> 4) & 31;
  int ks   = (idx >> 9) & 63;
  int nt   = idx >> 15;
  int K = ks * 32 + ((lane >= 16) ? 8 : 0) + (e & 7) + ((e & 8) << 1);       // 0..2047
  int N = nt * 16 + (lane & 15);                       // 0..511
  out[idx] = (__bf16)((N < H_) ? iHW[(size_t)K * H_ + N]
                               : iCW[(size_t)K * H_ + (N - H_)]);
}

__global__ void make_bcat(const float* bih, const float* bhh, float* bcat) {
  int i = blockIdx.x * blockDim.x + threadIdx.x;       // 1024
  bcat[i] = bih[i] + bhh[i];
}

// ---------------- init ------------------------------------------------------
__global__ void feat_to_x(const float* __restrict__ feat, __bf16* __restrict__ xcat) {
  int i = blockIdx.x * blockDim.x + threadIdx.x;       // B_*F_ = 262144
  int b = i >> 11, f = i & 2047;
  xcat[(size_t)b * KX + E_ + f] = (__bf16)feat[i];
}

__global__ void embed_to_x(const int* __restrict__ cap, const float* __restrict__ emb,
                           __bf16* __restrict__ xcat, int t) {
  int i = blockIdx.x * blockDim.x + threadIdx.x;       // B_*E_ = 16384
  int b = i >> 7, e = i & 127;
  int w = cap[b * T_ + t];
  xcat[(size_t)b * KX + e] = (__bf16)emb[(size_t)w * E_ + e];
}

// [h0|c0] = feat_bf16 @ packed_init  (128 x 512, K=2048). 256 waves.
__global__ void init_gemm(const __bf16* __restrict__ xcat_ro, const __bf16* __restrict__ Bp,
                          const float* __restrict__ iHb, const float* __restrict__ iCb,
                          float* __restrict__ cbuf, __bf16* __restrict__ xcat) {
  int wave = (blockIdx.x * blockDim.x + threadIdx.x) >> 5;   // 0..255
  int lane = threadIdx.x & 31;
  int mt = wave >> 5;            // 0..7
  int nt = wave & 31;            // 0..31
  const __bf16* abase = xcat_ro + E_
                      + (size_t)(mt * 16 + (lane & 15)) * KX + ((lane >> 4) << 3);
  const __bf16* bbase = Bp + ((size_t)(nt * KIS) * 32 + lane) * 16;
  v8f c = {};
  for (int k = 0; k < KIS; ++k) {
    __builtin_prefetch(bbase + (size_t)(k + 4) * 512, 0, 3);
    v16bf a = load_a(abase + k * 32);
    v16bf b = *(const v16bf*)(bbase + (size_t)k * 512);
    c = __builtin_amdgcn_wmma_f32_16x16x32_bf16(false, a, false, b, (short)0, c,
                                                false, false);
  }
  int n  = nt * 16 + (lane & 15);
  int m0 = mt * 16 + ((lane >> 4) << 3);
  if (n < H_) {                 // tile-aligned boundary -> wave-uniform branch
    float bias = iHb[n];
#pragma unroll
    for (int r = 0; r < 8; ++r)
      xcat[(size_t)(m0 + r) * KX + E_ + F_ + n] = (__bf16)(c[r] + bias);
  } else {
    int j = n - H_;
    float bias = iCb[j];
#pragma unroll
    for (int r = 0; r < 8; ++r)
      cbuf[(size_t)(m0 + r) * H_ + j] = c[r] + bias;
  }
}

// ---------------- per-step kernels ------------------------------------------
// gates partials: split-K x4, 2048 waves in ONE launch (4x the parallelism of
// one-wave-per-tile; each wave does only 19 load->wmma steps of exposed L2
// latency). Partial kq -> gbuf + kq*GBS.
__global__ void gates_gemm(const __bf16* __restrict__ xcat, const __bf16* __restrict__ Bp,
                           float* __restrict__ gbuf) {
  int wave = (blockIdx.x * blockDim.x + threadIdx.x) >> 5;   // 0..2047
  int lane = threadIdx.x & 31;
  int tile = wave & 511;
  int kq   = wave >> 9;          // 0..3
  int mt = tile >> 6;            // 0..7
  int nt = tile & 63;            // 0..63
  int k0 = kq * KCH;
  const __bf16* abase = xcat + (size_t)(mt * 16 + (lane & 15)) * KX
                      + ((lane >> 4) << 3) + k0 * 32;
  const __bf16* bbase = Bp + ((size_t)(nt * KXS + k0) * 32 + lane) * 16;
  v8f c = {};
  for (int k = 0; k < KCH; ++k) {
    __builtin_prefetch(bbase + (size_t)(k + 4) * 512, 0, 3);
    v16bf a = load_a(abase + k * 32);
    v16bf b = *(const v16bf*)(bbase + (size_t)k * 512);
    c = __builtin_amdgcn_wmma_f32_16x16x32_bf16(false, a, false, b, (short)0, c,
                                                false, false);
  }
  int n  = nt * 16 + (lane & 15);
  int m0 = mt * 16 + ((lane >> 4) << 3);
  float* g = gbuf + (size_t)kq * GBS;
#pragma unroll
  for (int r = 0; r < 8; ++r)
    g[(size_t)(m0 + r) * G4H + n] = c[r];
}

// sum 4 partials + bias, LSTM pointwise, h_new -> xcat (bf16),
// and fetch next step's embedding (fused to save a launch per step).
__global__ void lstm_pointwise(const float* __restrict__ gbuf, const float* __restrict__ bcat,
                               float* __restrict__ cbuf, __bf16* __restrict__ xcat,
                               const int* __restrict__ cap, const float* __restrict__ emb,
                               int t_next) {
  int i = blockIdx.x * blockDim.x + threadIdx.x;       // 32768
  int b = i >> 8, j = i & 255;
  const float* g = gbuf + (size_t)b * G4H;
#define GSUM(col) (g[(col)] + g[(col) + GBS] + g[(col) + 2*GBS] + g[(col) + 3*GBS] + bcat[(col)])
  float ig = 1.f / (1.f + __expf(-GSUM(j)));
  float fg = 1.f / (1.f + __expf(-GSUM(j + 256)));
  float gg = tanhf(GSUM(j + 512));
  float og = 1.f / (1.f + __expf(-GSUM(j + 768)));
#undef GSUM
  float cn = fg * cbuf[i] + ig * gg;
  cbuf[i] = cn;
  xcat[(size_t)b * KX + E_ + F_ + j] = (__bf16)(og * tanhf(cn));   // h_new
  if (t_next < T_ && j < E_) {                                     // embed t+1
    int w = cap[b * T_ + t_next];
    xcat[(size_t)b * KX + j] = (__bf16)emb[(size_t)w * E_ + j];
  }
}

// out_t = h_new @ fc_W + fc_b  (128 x 32000, K=256). One wave per
// (m-tile, 4 n-tiles); A for all K=256 held in 64 VGPRs. 4000 waves.
__global__ void fc_gemm(const __bf16* __restrict__ xcat, const __bf16* __restrict__ Bp,
                        const float* __restrict__ fcb, float* __restrict__ out, int t) {
  int wave = (blockIdx.x * blockDim.x + threadIdx.x) >> 5;   // 0..3999
  int lane = threadIdx.x & 31;
  int mt  = wave / 500;
  int grp = wave % 500;
  const __bf16* abase = xcat + E_ + F_
                      + (size_t)(mt * 16 + (lane & 15)) * KX + ((lane >> 4) << 3);
  v16bf a[KFS];
#pragma unroll
  for (int k = 0; k < KFS; ++k) a[k] = load_a(abase + k * 32);

  int lanN = lane & 15;
  int m0   = mt * 16 + ((lane >> 4) << 3);
#pragma unroll
  for (int q = 0; q < 4; ++q) {
    int nt = grp * 4 + q;
    const __bf16* bbase = Bp + ((size_t)(nt * KFS) * 32 + lane) * 16;
    v8f c = {};
#pragma unroll
    for (int k = 0; k < KFS; ++k) {
      v16bf b = *(const v16bf*)(bbase + (size_t)k * 512);
      c = __builtin_amdgcn_wmma_f32_16x16x32_bf16(false, a[k], false, b, (short)0, c,
                                                  false, false);
    }
    int n = nt * 16 + lanN;
    float bias = fcb[n];
    float* o = out + (size_t)m0 * T_ * V_ + (size_t)t * V_ + n;
#pragma unroll
    for (int r = 0; r < 8; ++r)
      o[(size_t)r * T_ * V_] = c[r] + bias;   // 64B/half-wave; L2 merges lines
  }
}

// ---------------------------------------------------------------------------
extern "C" void kernel_launch(void* const* d_in, const int* in_sizes, int n_in,
                              void* d_out, int out_size, void* d_ws, size_t ws_size,
                              hipStream_t stream) {
  const float* features = (const float*)d_in[0];
  const int*   captions = (const int*)  d_in[1];
  const float* embed    = (const float*)d_in[2];
  const float* W_ih     = (const float*)d_in[3];
  const float* b_ih     = (const float*)d_in[4];
  const float* W_hh     = (const float*)d_in[5];
  const float* b_hh     = (const float*)d_in[6];
  const float* fc_W     = (const float*)d_in[7];
  const float* fc_b     = (const float*)d_in[8];
  // d_in[9..14] = attention weights -- provably unused (softmax over size-1 axis)
  const float* initH_W  = (const float*)d_in[15];
  const float* initH_b  = (const float*)d_in[16];
  const float* initC_W  = (const float*)d_in[17];
  const float* initC_b  = (const float*)d_in[18];
  float* out = (float*)d_out;

  // workspace layout (all 256B-aligned)
  char* ws = (char*)d_ws;
  __bf16* pfc   = (__bf16*)(ws);                       // 16,384,000 B
  __bf16* pg    = (__bf16*)(ws + 16384000);            //  4,980,736 B
  __bf16* pinit = (__bf16*)(ws + 21364736);            //  2,097,152 B
  float*  bcat  = (float*) (ws + 23461888);            //      4,096 B
  __bf16* xcat  = (__bf16*)(ws + 23465984);            //    622,592 B
  float*  cbuf  = (float*) (ws + 24088576);            //    131,072 B
  float*  gbuf  = (float*) (ws + 24219648);            //  2,097,152 B (4 partials)

  // preprocessing (idempotent every call)
  pack_fc   <<<32000, 256, 0, stream>>>(fc_W, pfc);
  pack_gates<<< 9728, 256, 0, stream>>>(W_ih, W_hh, pg);
  pack_init <<< 4096, 256, 0, stream>>>(initH_W, initC_W, pinit);
  make_bcat <<<    4, 256, 0, stream>>>(b_ih, b_hh, bcat);
  feat_to_x <<< 1024, 256, 0, stream>>>(features, xcat);
  embed_to_x<<<   64, 256, 0, stream>>>(captions, embed, xcat, 0);
  init_gemm <<<   64, 128, 0, stream>>>(xcat, pinit, initH_b, initC_b, cbuf, xcat);

  for (int t = 0; t < T_; ++t) {
    gates_gemm    <<< 512, 128, 0, stream>>>(xcat, pg, gbuf);
    lstm_pointwise<<< 128, 256, 0, stream>>>(gbuf, bcat, cbuf, xcat,
                                             captions, embed, t + 1);
    fc_gemm       <<<1000, 128, 0, stream>>>(xcat, pfc, fc_b, out, t);
  }
}
